// SSMBackbone_21938692948378
// MI455X (gfx1250) — matrix operand order repaired and checked by reading, not compiled, not run on hardware
//
#include <hip/hip_runtime.h>
#include <cstddef>

// ---------------- model dims ----------------
constexpr int S      = 2048;   // sequence length (B=1)
constexpr int DIN    = 256;    // D_INPUT
constexpr int DM     = 1024;   // D_MODEL
constexpr int DI     = 2048;   // D_INNER
constexpr int DCONV  = 4;
constexpr int DSTATE = 16;
constexpr int DTR    = 65;     // DT_RANK
constexpr int DTRP   = 96;     // DT_RANK padded to K%32==0
constexpr int XDW    = 128;    // x_proj out width padded (97 -> 128, N%64==0)
constexpr int DFF    = 4096;
constexpr int NCH    = 16;     // scan chunks
constexpr int CHL    = S / NCH;

// ---------------- vector types ----------------
typedef __attribute__((ext_vector_type(16))) __bf16 v16bf;
typedef __attribute__((ext_vector_type(8)))  __bf16 v8bf;
typedef __attribute__((ext_vector_type(8)))  float  v8f;

__device__ __forceinline__ float softplus_f(float x) {
    return (x > 20.0f) ? x : log1pf(__expf(x));
}
__device__ __forceinline__ float silu_f(float x) {
    return x / (1.0f + __expf(-x));
}

// =====================================================================
// fp32 -> bf16 convert with zero padding in both dims (also plain cvt)
// =====================================================================
__global__ void cvt_pad_bf16(const float* __restrict__ src, __bf16* __restrict__ dst,
                             int srcRows, int srcCols, int srcStride,
                             int dstRows, int dstCols)
{
    int idx = blockIdx.x * blockDim.x + threadIdx.x;
    int total = dstRows * dstCols;
    if (idx >= total) return;
    int r = idx / dstCols, c = idx - r * dstCols;
    float v = (r < srcRows && c < srcCols) ? src[(size_t)r * srcStride + c] : 0.0f;
    dst[idx] = (__bf16)v;
}

// =====================================================================
// bf16 WMMA GEMM:  out[M,N](f32) = A[M,K] * W[N,K]^T (+bias[N]) (+res[M,N])
// 8 waves / block, wave tile = 32(M) x 64(N): 8 x v_wmma_f32_16x16x32_bf16
// per K-step against 12 b128 loads.  A/B both use contiguous k = 16*half + s per lane.
// Requires K%32==0, N%64==0, M%256==0 (grid), row stride of out/res == N.
// =====================================================================
__global__ __launch_bounds__(256)
void gemm_bf16_wmma(const __bf16* __restrict__ A, const __bf16* __restrict__ W,
                    const float* __restrict__ bias, const float* __restrict__ res,
                    float* __restrict__ out, int M, int N, int K)
{
    const int wave = threadIdx.x >> 5;   // wave32
    const int lane = threadIdx.x & 31;
    const int half = lane >> 4;          // 0: lanes 0-15, 1: lanes 16-31
    const int l16  = lane & 15;

    const int m0 = blockIdx.y * 256 + wave * 32;
    const int n0 = blockIdx.x * 64;
    if (m0 >= M) return;

    v8f acc[2][4] = {};

    // A: lane l16 = row M, K-halves by lane half
    const __bf16* Arow0 = A + (size_t)(m0      + l16) * K;
    const __bf16* Arow1 = A + (size_t)(m0 + 16 + l16) * K;
    // B: lane l16 = column N, 16 contiguous K per lane half
    const __bf16* Wc0 = W + (size_t)(n0      + l16) * K + half * 16;
    const __bf16* Wc1 = W + (size_t)(n0 + 16 + l16) * K + half * 16;
    const __bf16* Wc2 = W + (size_t)(n0 + 32 + l16) * K + half * 16;
    const __bf16* Wc3 = W + (size_t)(n0 + 48 + l16) * K + half * 16;

    for (int k = 0; k < K; k += 32) {
        v16bf a0, a1;
        {
            v8bf lo = *(const v8bf*)(Arow0 + k +     half * 16);
            v8bf hi = *(const v8bf*)(Arow0 + k + 8 + half * 16);
#pragma unroll
            for (int i = 0; i < 8; ++i) { a0[i] = lo[i]; a0[8 + i] = hi[i]; }
        }
        {
            v8bf lo = *(const v8bf*)(Arow1 + k +     half * 16);
            v8bf hi = *(const v8bf*)(Arow1 + k + 8 + half * 16);
#pragma unroll
            for (int i = 0; i < 8; ++i) { a1[i] = lo[i]; a1[8 + i] = hi[i]; }
        }

        v16bf b0 = *(const v16bf*)(Wc0 + k);
        v16bf b1 = *(const v16bf*)(Wc1 + k);
        v16bf b2 = *(const v16bf*)(Wc2 + k);
        v16bf b3 = *(const v16bf*)(Wc3 + k);

        if (k + 32 < K) {                       // stream next weight tiles
            __builtin_prefetch(Wc0 + k + 32, 0, 1);
            __builtin_prefetch(Wc2 + k + 32, 0, 1);
        }

        acc[0][0] = __builtin_amdgcn_wmma_f32_16x16x32_bf16(false, a0, false, b0, (short)0, acc[0][0], false, false);
        acc[0][1] = __builtin_amdgcn_wmma_f32_16x16x32_bf16(false, a0, false, b1, (short)0, acc[0][1], false, false);
        acc[0][2] = __builtin_amdgcn_wmma_f32_16x16x32_bf16(false, a0, false, b2, (short)0, acc[0][2], false, false);
        acc[0][3] = __builtin_amdgcn_wmma_f32_16x16x32_bf16(false, a0, false, b3, (short)0, acc[0][3], false, false);
        acc[1][0] = __builtin_amdgcn_wmma_f32_16x16x32_bf16(false, a1, false, b0, (short)0, acc[1][0], false, false);
        acc[1][1] = __builtin_amdgcn_wmma_f32_16x16x32_bf16(false, a1, false, b1, (short)0, acc[1][1], false, false);
        acc[1][2] = __builtin_amdgcn_wmma_f32_16x16x32_bf16(false, a1, false, b2, (short)0, acc[1][2], false, false);
        acc[1][3] = __builtin_amdgcn_wmma_f32_16x16x32_bf16(false, a1, false, b3, (short)0, acc[1][3], false, false);
    }

    // D layout: lane l16 = column, VGPR r = row (half selects M 0-7 vs 8-15)
#pragma unroll
    for (int ms = 0; ms < 2; ++ms) {
        const int row0 = m0 + ms * 16 + half * 8;
#pragma unroll
        for (int nt = 0; nt < 4; ++nt) {
            const int col = n0 + nt * 16 + l16;
            const float bv = bias ? bias[col] : 0.0f;
#pragma unroll
            for (int r = 0; r < 8; ++r) {
                size_t o = (size_t)(row0 + r) * N + col;
                float v = acc[ms][nt][r] + bv;
                if (res) v += res[o];
                out[o] = v;
            }
        }
    }
}

// =====================================================================
// LayerNorm over D=1024, one 256-thread block per row.
// =====================================================================
__global__ __launch_bounds__(256)
void layernorm_k(const float* __restrict__ x, const float* __restrict__ w,
                 const float* __restrict__ b, __bf16* __restrict__ obf,
                 float* __restrict__ of32)
{
    __shared__ float red[256];
    const int row = blockIdx.x;
    const float* xr = x + (size_t)row * DM;

    float vals[4];
    float s = 0.0f;
#pragma unroll
    for (int i = 0; i < 4; ++i) { vals[i] = xr[threadIdx.x + i * 256]; s += vals[i]; }
    red[threadIdx.x] = s; __syncthreads();
    for (int o = 128; o > 0; o >>= 1) {
        if (threadIdx.x < o) red[threadIdx.x] += red[threadIdx.x + o];
        __syncthreads();
    }
    const float mean = red[0] * (1.0f / DM);
    __syncthreads();

    float v = 0.0f;
#pragma unroll
    for (int i = 0; i < 4; ++i) { float d = vals[i] - mean; v += d * d; }
    red[threadIdx.x] = v; __syncthreads();
    for (int o = 128; o > 0; o >>= 1) {
        if (threadIdx.x < o) red[threadIdx.x] += red[threadIdx.x + o];
        __syncthreads();
    }
    const float rstd = rsqrtf(red[0] * (1.0f / DM) + 1e-5f);

#pragma unroll
    for (int i = 0; i < 4; ++i) {
        const int c = threadIdx.x + i * 256;
        const float y = (vals[i] - mean) * rstd * w[c] + b[c];
        if (obf)  obf[(size_t)row * DM + c] = (__bf16)y;
        if (of32) of32[(size_t)row * DM + c] = y;
    }
}

// =====================================================================
// Depthwise causal conv (D_CONV=4) + bias + SiLU.
// =====================================================================
__global__ __launch_bounds__(256)
void conv_silu_k(const float* __restrict__ XZ, const float* __restrict__ cw,
                 const float* __restrict__ cb, float* __restrict__ XC,
                 __bf16* __restrict__ XCbf)
{
    const int idx = blockIdx.x * blockDim.x + threadIdx.x;   // over S*DI
    const int t = idx / DI, d = idx - t * DI;
    float acc = cb[d];
#pragma unroll
    for (int kk = 0; kk < DCONV; ++kk) {
        const int tt = t + kk - (DCONV - 1);
        if (tt >= 0) acc += XZ[(size_t)tt * (2 * DI) + d] * cw[d * DCONV + kk];
    }
    const float sv = silu_f(acc);
    XC[idx] = sv;
    XCbf[idx] = (__bf16)sv;
}

// =====================================================================
// Chunked selective scan (linear recurrence s_t = a_t*s_{t-1} + b_t):
//   phase1: per (chunk,d) compute chunk decay product P and local result Q
//   phase2: per d, sequential combine over 16 chunks -> chunk start states
//   phase3: per (chunk,d) replay with correct s0; emit gated bf16 y
// 16x more parallelism than a monolithic scan for ~2x the (cheap) FLOPs.
// =====================================================================
__global__ __launch_bounds__(256)
void scan_phase1(const float* __restrict__ DTraw, const float* __restrict__ XC,
                 const float* __restrict__ XDBL, const float* __restrict__ A_log,
                 float* __restrict__ P, float* __restrict__ Q)
{
    const int gid = blockIdx.x * blockDim.x + threadIdx.x;   // over NCH*DI
    if (gid >= NCH * DI) return;
    const int c = gid / DI, d = gid - c * DI;

    float A[DSTATE], p[DSTATE], q[DSTATE];
#pragma unroll
    for (int n = 0; n < DSTATE; ++n) {
        A[n] = -__expf(A_log[d * DSTATE + n]);
        p[n] = 1.0f; q[n] = 0.0f;
    }
    const int t0 = c * CHL;
    for (int tt = 0; tt < CHL; ++tt) {
        const int t = t0 + tt;
        const float dt = softplus_f(DTraw[(size_t)t * DI + d]);
        const float dx = dt * XC[(size_t)t * DI + d];
        const float* xd = XDBL + (size_t)t * XDW;
#pragma unroll
        for (int n = 0; n < DSTATE; ++n) {
            const float e = __expf(dt * A[n]);
            p[n] *= e;
            q[n] = e * q[n] + dx * xd[DTR + n];
        }
    }
    const size_t base = ((size_t)c * DI + d) * DSTATE;
#pragma unroll
    for (int n = 0; n < DSTATE; ++n) { P[base + n] = p[n]; Q[base + n] = q[n]; }
}

__global__ __launch_bounds__(256)
void scan_phase2(const float* __restrict__ P, const float* __restrict__ Q,
                 float* __restrict__ S0)
{
    const int d = blockIdx.x * blockDim.x + threadIdx.x;
    if (d >= DI) return;
    float s[DSTATE];
#pragma unroll
    for (int n = 0; n < DSTATE; ++n) s[n] = 0.0f;
    for (int c = 0; c < NCH; ++c) {
        const size_t base = ((size_t)c * DI + d) * DSTATE;
#pragma unroll
        for (int n = 0; n < DSTATE; ++n) {
            S0[base + n] = s[n];                       // state entering chunk c
            s[n] = P[base + n] * s[n] + Q[base + n];
        }
    }
}

__global__ __launch_bounds__(256)
void scan_phase3(const float* __restrict__ DTraw, const float* __restrict__ XC,
                 const float* __restrict__ XZ, const float* __restrict__ XDBL,
                 const float* __restrict__ A_log, const float* __restrict__ D_skip,
                 const float* __restrict__ S0, __bf16* __restrict__ Ybf)
{
    const int gid = blockIdx.x * blockDim.x + threadIdx.x;   // over NCH*DI
    if (gid >= NCH * DI) return;
    const int c = gid / DI, d = gid - c * DI;

    float A[DSTATE], s[DSTATE];
    const size_t base = ((size_t)c * DI + d) * DSTATE;
#pragma unroll
    for (int n = 0; n < DSTATE; ++n) {
        A[n] = -__expf(A_log[d * DSTATE + n]);
        s[n] = S0[base + n];
    }
    const float dsk = D_skip[d];

    const int t0 = c * CHL;
    for (int tt = 0; tt < CHL; ++tt) {
        const int t = t0 + tt;
        const float dt = softplus_f(DTraw[(size_t)t * DI + d]);
        const float xv = XC[(size_t)t * DI + d];
        const float zv = XZ[(size_t)t * (2 * DI) + DI + d];
        const float* xd = XDBL + (size_t)t * XDW;
        const float dx = dt * xv;
        float y = 0.0f;
#pragma unroll
        for (int n = 0; n < DSTATE; ++n) {
            s[n] = __expf(dt * A[n]) * s[n] + dx * xd[DTR + n];
            y += s[n] * xd[DTR + DSTATE + n];
        }
        y += xv * dsk;
        y *= silu_f(zv);
        Ybf[(size_t)t * DI + d] = (__bf16)y;
    }
}

// =====================================================================
// MLP gate: out_bf16 = SiLU(G) * U
// =====================================================================
__global__ __launch_bounds__(256)
void glu_k(const float* __restrict__ G, const float* __restrict__ U,
           __bf16* __restrict__ Gbf, int n)
{
    const int idx = blockIdx.x * blockDim.x + threadIdx.x;
    if (idx >= n) return;
    Gbf[idx] = (__bf16)(silu_f(G[idx]) * U[idx]);
}

// =====================================================================
// host launcher
// =====================================================================
extern "C" void kernel_launch(void* const* d_in, const int* in_sizes, int n_in,
                              void* d_out, int out_size, void* d_ws, size_t ws_size,
                              hipStream_t stream)
{
    (void)in_sizes; (void)n_in; (void)out_size; (void)ws_size;

    const float* x            = (const float*)d_in[0];
    const float* input_proj_w = (const float*)d_in[1];
    const float* input_proj_b = (const float*)d_in[2];
    const float* m_norm_w     = (const float*)d_in[3];
    const float* m_norm_b     = (const float*)d_in[4];
    const float* in_proj_w    = (const float*)d_in[5];
    const float* conv_w       = (const float*)d_in[6];
    const float* conv_b       = (const float*)d_in[7];
    const float* x_proj_w     = (const float*)d_in[8];
    const float* dt_proj_w    = (const float*)d_in[9];
    const float* dt_proj_b    = (const float*)d_in[10];
    const float* A_log        = (const float*)d_in[11];
    const float* D_skip       = (const float*)d_in[12];
    const float* out_proj_w   = (const float*)d_in[13];
    const float* p_norm_w     = (const float*)d_in[14];
    const float* p_norm_b     = (const float*)d_in[15];
    const float* gate_w       = (const float*)d_in[16];
    const float* up_w         = (const float*)d_in[17];
    const float* down_w       = (const float*)d_in[18];
    const float* fn_w         = (const float*)d_in[19];
    const float* fn_b         = (const float*)d_in[20];

    // ---- bump allocator over workspace (256B aligned) ----
    char* ws = (char*)d_ws;
    size_t off = 0;
    auto alloc = [&](size_t bytes) -> void* {
        off = (off + 255) & ~(size_t)255;
        void* p = ws + off;
        off += bytes;
        return p;
    };

    float*  H0    = (float*) alloc((size_t)S * DM * 4);
    float*  H1    = (float*) alloc((size_t)S * DM * 4);
    __bf16* hnbf  = (__bf16*)alloc((size_t)S * DM * 2);
    __bf16* wbf0  = (__bf16*)alloc((size_t)DFF * DM * 2);   // largest weight: 4096x1024
    __bf16* wbf1  = (__bf16*)alloc((size_t)DFF * DM * 2);
    __bf16* xbf   = (__bf16*)alloc((size_t)S * DIN * 2);
    float*  XZ    = (float*) alloc((size_t)S * 2 * DI * 4);
    float*  XC    = (float*) alloc((size_t)S * DI * 4);
    __bf16* XCbf  = (__bf16*)alloc((size_t)S * DI * 2);
    float*  XDBL  = (float*) alloc((size_t)S * XDW * 4);
    __bf16* dtrbf = (__bf16*)alloc((size_t)S * DTRP * 2);
    float*  DT    = (float*) alloc((size_t)S * DI * 4);
    __bf16* Ybf   = (__bf16*)alloc((size_t)S * DI * 2);
    float*  G     = (float*) alloc((size_t)S * DFF * 4);
    float*  U     = (float*) alloc((size_t)S * DFF * 4);
    __bf16* Gbf   = (__bf16*)alloc((size_t)S * DFF * 2);
    float*  Pbuf  = (float*) alloc((size_t)NCH * DI * DSTATE * 4);
    float*  Qbuf  = (float*) alloc((size_t)NCH * DI * DSTATE * 4);
    float*  S0buf = (float*) alloc((size_t)NCH * DI * DSTATE * 4);

    auto cvt = [&](const float* s, __bf16* dptr, int sr, int sc, int sstr, int dr, int dc) {
        int total = dr * dc;
        cvt_pad_bf16<<<(total + 255) / 256, 256, 0, stream>>>(s, dptr, sr, sc, sstr, dr, dc);
    };
    auto gemm = [&](const __bf16* A, const __bf16* W, const float* bias,
                    const float* res, float* out, int M, int N, int K) {
        dim3 g(N / 64, (M + 255) / 256);
        gemm_bf16_wmma<<<g, 256, 0, stream>>>(A, W, bias, res, out, M, N, K);
    };

    // ---- input projection: H0 = x @ input_proj_w^T + b ----
    cvt(x,            xbf,  S,  DIN, DIN, S,  DIN);
    cvt(input_proj_w, wbf0, DM, DIN, DIN, DM, DIN);
    gemm(xbf, wbf0, input_proj_b, nullptr, H0, S, DM, DIN);

    float* Hc = H0;
    float* Hn = H1;

    // ---- two Mamba blocks ----
    for (int k = 0; k < 2; ++k) {
        layernorm_k<<<S, 256, 0, stream>>>(Hc, m_norm_w + k * DM, m_norm_b + k * DM, hnbf, nullptr);

        cvt(in_proj_w + (size_t)k * 2 * DI * DM, wbf0, 2 * DI, DM, DM, 2 * DI, DM);
        gemm(hnbf, wbf0, nullptr, nullptr, XZ, S, 2 * DI, DM);

        conv_silu_k<<<(S * DI) / 256, 256, 0, stream>>>(XZ, conv_w + (size_t)k * DI * DCONV,
                                                        conv_b + (size_t)k * DI, XC, XCbf);

        cvt(x_proj_w + (size_t)k * (DTR + 2 * DSTATE) * DI, wbf0,
            DTR + 2 * DSTATE, DI, DI, XDW, DI);               // pad 97 -> 128 rows
        gemm(XCbf, wbf0, nullptr, nullptr, XDBL, S, XDW, DI);

        cvt(XDBL, dtrbf, S, DTR, XDW, S, DTRP);               // dt_r slice, pad 65 -> 96
        cvt(dt_proj_w + (size_t)k * DI * DTR, wbf0, DI, DTR, DTR, DI, DTRP);
        gemm(dtrbf, wbf0, dt_proj_b + (size_t)k * DI, nullptr, DT, S, DI, DTRP);

        const float* Ak = A_log + (size_t)k * DI * DSTATE;
        scan_phase1<<<(NCH * DI) / 256, 256, 0, stream>>>(DT, XC, XDBL, Ak, Pbuf, Qbuf);
        scan_phase2<<<DI / 256, 256, 0, stream>>>(Pbuf, Qbuf, S0buf);
        scan_phase3<<<(NCH * DI) / 256, 256, 0, stream>>>(DT, XC, XZ, XDBL, Ak,
                                                          D_skip + (size_t)k * DI, S0buf, Ybf);

        cvt(out_proj_w + (size_t)k * DM * DI, wbf0, DM, DI, DI, DM, DI);
        gemm(Ybf, wbf0, nullptr, Hc, Hn, S, DM, DI);          // fused residual add
        float* tmp = Hc; Hc = Hn; Hn = tmp;
    }

    // ---- MLP block ----
    layernorm_k<<<S, 256, 0, stream>>>(Hc, p_norm_w, p_norm_b, hnbf, nullptr);
    cvt(gate_w, wbf0, DFF, DM, DM, DFF, DM);
    cvt(up_w,   wbf1, DFF, DM, DM, DFF, DM);
    gemm(hnbf, wbf0, nullptr, nullptr, G, S, DFF, DM);
    gemm(hnbf, wbf1, nullptr, nullptr, U, S, DFF, DM);
    glu_k<<<(S * DFF) / 256, 256, 0, stream>>>(G, U, Gbf, S * DFF);
    cvt(down_w, wbf0, DM, DFF, DFF, DM, DFF);
    gemm(Gbf, wbf0, nullptr, Hc, Hn, S, DM, DFF);             // fused residual add
    { float* tmp = Hc; Hc = Hn; Hn = tmp; }

    // ---- final LayerNorm -> fp32 output ----
    layernorm_k<<<S, 256, 0, stream>>>(Hc, fn_w, fn_b, nullptr, (float*)d_out);
}
